// DeepLSTM_317827580263
// MI455X (gfx1250) — compile-verified, matrix operand order checked by reading, not run on hardware
//
#include <hip/hip_runtime.h>
#include <cstdint>

#define D_ 512
#define H_ 1024
#define B_ 128
#define T_ 256

typedef __attribute__((ext_vector_type(16))) __bf16       v16bf;
typedef __attribute__((ext_vector_type(8)))  float        v8f;

// GCC-vector types matching the async builtin's prototype
typedef int v4i __attribute__((vector_size(16)));
typedef int v8i_g __attribute__((vector_size(32)));
typedef __attribute__((address_space(1))) v4i*       gv4i_ptr;
typedef __attribute__((address_space(3))) v4i*       lv4i_ptr;
typedef __attribute__((address_space(3))) const v4i* lv4i_cptr;

// LDS row pitch for 32 bf16 + pad, in elements (80 B, multiple of 16 B)
#define PITCH 40

// ---- CDNA5 async global->LDS path (guarded; falls back to VGPR staging) ----
#if defined(__has_builtin)
#if __has_builtin(__builtin_amdgcn_global_load_async_to_lds_b128) && \
    __has_builtin(__builtin_amdgcn_s_wait_asynccnt)
#define USE_ASYNC 1
#endif
#endif
#ifndef USE_ASYNC
#define USE_ASYNC 0
#endif

__device__ __forceinline__ void cp16_issue(const __bf16* g, __bf16* l, uint4* reg) {
#if USE_ASYNC
  __builtin_amdgcn_global_load_async_to_lds_b128(
      (gv4i_ptr)(uintptr_t)g,
      (lv4i_ptr)(unsigned)(uintptr_t)l,
      0, 0);
  (void)reg;
#else
  *reg = *(const uint4*)g;
  (void)l;
#endif
}

__device__ __forceinline__ void cp16_commit(__bf16* l, const uint4* reg) {
#if !USE_ASYNC
  *(lv4i_ptr)(unsigned)(uintptr_t)l = __builtin_bit_cast(v4i, *reg);
#else
  (void)l; (void)reg;
#endif
}

__device__ __forceinline__ void tile_wait() {
#if USE_ASYNC
  __builtin_amdgcn_s_wait_asynccnt(0);
#endif
}

// 16-byte LDS read forced down the DS path (addrspace(3) pointer, not flat)
__device__ __forceinline__ v4i ld_lds16(const __bf16* l) {
  return *(lv4i_cptr)(unsigned)(uintptr_t)l;
}

// ===================== prep kernels =====================

__global__ __launch_bounds__(256)
void pack_x_kernel(const float* __restrict__ x, __bf16* __restrict__ xbf) {
  const long total = (long)B_ * T_ * D_;
  for (long i = (long)blockIdx.x * 256 + threadIdx.x; i < total; i += (long)gridDim.x * 256) {
    int d   = (int)(i % D_);
    long bt = i / D_;
    int b   = (int)(bt % B_);
    int t   = (int)(bt / B_);
    xbf[i]  = (__bf16)x[((long)b * T_ + t) * D_ + d];   // out layout [t][b][d]
  }
}

__global__ __launch_bounds__(256)
void pack_weights_kernel(const float* __restrict__ A, int Ka,
                         const float* __restrict__ Bs, int Kb,
                         __bf16* __restrict__ dst, long total) {
  const int K = Ka + Kb;
  for (long i = (long)blockIdx.x * 256 + threadIdx.x; i < total; i += (long)gridDim.x * 256) {
    long n = i / K;
    int  k = (int)(i % K);
    float v = (k < Ka) ? A[n * (long)Ka + k] : Bs[n * (long)Kb + (k - Ka)];
    dst[i] = (__bf16)v;
  }
}

__global__ __launch_bounds__(256)
void pack_bias_kernel(const float* __restrict__ a, const float* __restrict__ b,
                      float* __restrict__ dst, int n, int hasB) {
  for (int i = blockIdx.x * 256 + threadIdx.x; i < n; i += gridDim.x * 256)
    dst[i] = a[i] + (hasB ? b[i] : 0.f);
}

__global__ __launch_bounds__(256)
void zero_state_kernel(__bf16* __restrict__ h, float* __restrict__ c, long nh, long nc) {
  for (long i = (long)blockIdx.x * 256 + threadIdx.x; i < nh; i += (long)gridDim.x * 256)
    h[i] = (__bf16)0.f;
  for (long i = (long)blockIdx.x * 256 + threadIdx.x; i < nc; i += (long)gridDim.x * 256)
    c[i] = 0.f;
}

// ===================== tile staging =====================
// A tile: 128 rows x 32 k bf16 (8 KB) -> 512 16-byte chunks, 2 per thread.
// B tile: NT*16 rows x 32 k          -> NT*64 chunks.
template <int NT>
__device__ __forceinline__ void tile_issue(const __bf16* __restrict__ ab, long astr, int aoff,
                                           const __bf16* __restrict__ W, long Wld, int kbg, int j0,
                                           __bf16* smA, __bf16* smB, uint4* rA, uint4* rB) {
  const int tid = threadIdx.x;
#pragma unroll
  for (int rep = 0; rep < 2; ++rep) {
    int c0 = tid + rep * 256;
    int m = c0 >> 2, part = c0 & 3;
    cp16_issue(&ab[(long)m * astr + aoff + part * 8], &smA[m * PITCH + part * 8], &rA[rep]);
  }
#pragma unroll
  for (int rep = 0; rep < (NT * 64 + 255) / 256; ++rep) {
    int c = tid + rep * 256;
    if (c < NT * 64) {
      int nlr = c >> 2, part = c & 3;
      int grow = (nlr >> 5) * 1024 + j0 + (nlr & 31);
      cp16_issue(&W[(long)grow * Wld + kbg + part * 8], &smB[nlr * PITCH + part * 8], &rB[rep]);
    }
  }
}

template <int NT>
__device__ __forceinline__ void tile_commit(__bf16* smA, __bf16* smB, uint4* rA, uint4* rB) {
#if !USE_ASYNC
  const int tid = threadIdx.x;
#pragma unroll
  for (int rep = 0; rep < 2; ++rep) {
    int c0 = tid + rep * 256;
    int m = c0 >> 2, part = c0 & 3;
    cp16_commit(&smA[m * PITCH + part * 8], &rA[rep]);
  }
#pragma unroll
  for (int rep = 0; rep < (NT * 64 + 255) / 256; ++rep) {
    int c = tid + rep * 256;
    if (c < NT * 64)
      cp16_commit(&smB[(c >> 2) * PITCH + (c & 3) * 8], &rB[rep]);
  }
#else
  (void)smA; (void)smB; (void)rA; (void)rB;
#endif
}

// ===================== fragment compute =====================
__device__ __forceinline__ v16bf pack_frag(v4i lo, v4i hi) {
  v8i_g u = {lo[0], lo[1], lo[2], lo[3], hi[0], hi[1], hi[2], hi[3]};
  return __builtin_bit_cast(v16bf, u);
}

// Per wave: one 16x32 A fragment, NT B fragments -> NT back-to-back WMMAs.
template <int NT>
__device__ __forceinline__ void compute_tile(const __bf16* smA, const __bf16* smB, v8f* acc) {
  const int lane = threadIdx.x & 31;
  const int grp = lane >> 4, nl = lane & 15;          // ISA bf16 layout: hi lanes k+8
  const int arow = (threadIdx.x >> 5) * 16 + nl;
  v16bf af = pack_frag(ld_lds16(&smA[arow * PITCH + grp * 8]),
                       ld_lds16(&smA[arow * PITCH + 16 + grp * 8]));
#pragma unroll
  for (int g = 0; g < NT; g += 4) {
    v16bf bfr[4];
#pragma unroll
    for (int c = 0; c < 4; ++c)
      if (g + c < NT) {
        int brow = (g + c) * 16 + nl;
        bfr[c] = pack_frag(ld_lds16(&smB[brow * PITCH + grp * 8]),
                           ld_lds16(&smB[brow * PITCH + 16 + grp * 8]));
      }
#pragma unroll
    for (int c = 0; c < 4; ++c)
      if (g + c < NT)
        acc[g + c] = __builtin_amdgcn_wmma_f32_16x16x32_bf16(
            false, af, false, bfr[c], (short)0, acc[g + c], false, false);
  }
}

// ===================== double-buffered GEMM pipeline =====================
// acc[m][nl] = sum_k A[m][k] * W[grow(nl)][k], A = [a0|a1|a2] split at k1,k2.
// Per-segment loops: no address select in the hot path (scalar-uniform control).
template <int NT>
__device__ __forceinline__ void gemm_pipeline(
    const __bf16* __restrict__ a0, const __bf16* __restrict__ a1,
    const __bf16* __restrict__ a2, int k1, int k2, int Ktot,
    const __bf16* __restrict__ W, long Wld, int j0, v8f* acc,
    __bf16* smA0, __bf16* smA1, __bf16* smB0, __bf16* smB1) {
  const __bf16* sb[3] = {a0, a1, a2};
  const long sstr[3]  = {(long)D_, (long)H_, (long)H_};
  const int  slen[3]  = {k1, k2 - k1, Ktot - k2};
  __bf16* bufA[2] = {smA0, smA1};
  __bf16* bufB[2] = {smB0, smB1};
  uint4 rA[2], rB[2];

  // prologue: stage first tile
  tile_issue<NT>(sb[0], sstr[0], 0, W, Wld, 0, j0, bufA[0], bufB[0], rA, rB);
  tile_commit<NT>(bufA[0], bufB[0], rA, rB);
  tile_wait();
  __syncthreads();

  int cur = 0, kbg = 0;
#pragma unroll 1
  for (int s = 0; s < 3; ++s) {
#pragma unroll 1
    for (int kb = 0; kb < slen[s]; kb += 32) {
      // next tile coordinates (skip empty segments)
      int ns = s, nkb = kb + 32;
      if (nkb >= slen[s]) {
        nkb = 0; ns = s + 1;
        if (ns < 3 && slen[ns] == 0) ++ns;
        if (ns < 3 && slen[ns] == 0) ++ns;
      }
      const bool have_next = (kbg + 32 < Ktot);
      if (have_next)
        tile_issue<NT>(sb[ns], sstr[ns], nkb, W, Wld, kbg + 32, j0,
                       bufA[cur ^ 1], bufB[cur ^ 1], rA, rB);
      compute_tile<NT>(bufA[cur], bufB[cur], acc);
      if (have_next) {
        tile_commit<NT>(bufA[cur ^ 1], bufB[cur ^ 1], rA, rB);
        tile_wait();
      }
      __syncthreads();
      cur ^= 1;
      kbg += 32;
    }
  }
}

// ===================== fused LSTM cell =====================
// gates[128][4096] = [x_t | h_low | h_self] @ Wcat^T, then i,f,g,o nonlinearity.
// WG covers hidden units [j0, j0+32): acc tiles {2g,2g+1} are gate g.
__global__ __launch_bounds__(256)
void lstm_cell_kernel(const __bf16* __restrict__ xt,
                      const __bf16* __restrict__ hlow,
                      const __bf16* __restrict__ hself,
                      const __bf16* __restrict__ W, int Ktot, int k2,
                      const float* __restrict__ bias,
                      float* __restrict__ c_st,
                      __bf16* __restrict__ h_out) {
  __shared__ __bf16 smA0[128 * PITCH], smA1[128 * PITCH];
  __shared__ __bf16 smB0[128 * PITCH], smB1[128 * PITCH];
  v8f acc[8];
#pragma unroll
  for (int c = 0; c < 8; ++c)
#pragma unroll
    for (int r = 0; r < 8; ++r) acc[c][r] = 0.f;

  const int j0 = blockIdx.x * 32;
  gemm_pipeline<8>(xt, hlow, hself, D_, k2, Ktot, W, (long)Ktot, j0, acc,
                   smA0, smA1, smB0, smB1);

  const int lane = threadIdx.x & 31, wave = threadIdx.x >> 5;
  const int grp = lane >> 4, nl = lane & 15;
#pragma unroll
  for (int c2 = 0; c2 < 2; ++c2) {
    const int u = j0 + c2 * 16 + nl;
    const float bi = bias[u], bff = bias[1024 + u], bg = bias[2048 + u], bo = bias[3072 + u];
#pragma unroll
    for (int r = 0; r < 8; ++r) {
      const int m = wave * 16 + grp * 8 + r;   // C/D layout: VGPR r -> M=r (+8 hi lanes)
      const long idx = (long)m * H_ + u;
      float iv = acc[0 + c2][r] + bi;
      float fv = acc[2 + c2][r] + bff;
      float gv = acc[4 + c2][r] + bg;
      float ov = acc[6 + c2][r] + bo;
      float ig = 1.f / (1.f + __expf(-iv));
      float fg = 1.f / (1.f + __expf(-fv));
      float og = 1.f / (1.f + __expf(-ov));
      float gg = tanhf(gv);
      float cn = fg * c_st[idx] + ig * gg;
      c_st[idx] = cn;
      h_out[idx] = (__bf16)(og * tanhf(cn));
    }
  }
}

// ===================== output head GEMM (logits) =====================
__global__ __launch_bounds__(256)
void head_kernel(const __bf16* __restrict__ xt, const __bf16* __restrict__ h3,
                 const __bf16* __restrict__ Wlin, const float* __restrict__ blin,
                 float* __restrict__ logits) {
  __shared__ __bf16 smA0[128 * PITCH], smA1[128 * PITCH];
  __shared__ __bf16 smB0[32 * PITCH], smB1[32 * PITCH];
  v8f acc[2];
#pragma unroll
  for (int c = 0; c < 2; ++c)
#pragma unroll
    for (int r = 0; r < 8; ++r) acc[c][r] = 0.f;

  const int j0 = blockIdx.x * 32;
  gemm_pipeline<2>(xt, h3, h3, D_, D_ + H_, D_ + H_, Wlin, (long)(D_ + H_), j0, acc,
                   smA0, smA1, smB0, smB1);

  const int lane = threadIdx.x & 31, wave = threadIdx.x >> 5;
  const int grp = lane >> 4, nl = lane & 15;
#pragma unroll
  for (int c2 = 0; c2 < 2; ++c2) {
    const int col = j0 + c2 * 16 + nl;
    const float bb = blin[col];
#pragma unroll
    for (int r = 0; r < 8; ++r) {
      const int m = wave * 16 + grp * 8 + r;
      logits[(long)m * D_ + col] = acc[c2][r] + bb;
    }
  }
}

// ===================== softmax + store to d_out =====================
__global__ __launch_bounds__(256)
void softmax_kernel(const float* __restrict__ logits, float* __restrict__ out, int t) {
  __shared__ float red[256];
  const int b = blockIdx.x;
  const int tx = threadIdx.x;
  float v0 = logits[(long)b * D_ + tx];
  float v1 = logits[(long)b * D_ + 256 + tx];
  red[tx] = fmaxf(v0, v1);
  __syncthreads();
  for (int s = 128; s > 0; s >>= 1) {
    if (tx < s) red[tx] = fmaxf(red[tx], red[tx + s]);
    __syncthreads();
  }
  const float mx = red[0];
  __syncthreads();
  float e0 = __expf(v0 - mx), e1 = __expf(v1 - mx);
  red[tx] = e0 + e1;
  __syncthreads();
  for (int s = 128; s > 0; s >>= 1) {
    if (tx < s) red[tx] += red[tx + s];
    __syncthreads();
  }
  const float inv = 1.f / red[0];
  const long base = ((long)b * T_ + t) * D_;
  out[base + tx] = e0 * inv;
  out[base + 256 + tx] = e1 * inv;
}

// ===================== workspace layout (bytes) =====================
#define OFF_XBF   0L                         // 32 MB  bf16 x [T][B][D]
#define OFF_W0    33554432L                  // 12 MB  bf16 [4096][1536]
#define OFF_WR    46137344L                  // 60 MB  bf16 3x[4096][2560]
#define OFF_WL    109051904L                 // 1.5 MB bf16 [512][1536]
#define OFF_B0    110624768L                 // 16 KB  f32 [4096]
#define OFF_BR    110641152L                 // 48 KB  f32 3x[4096]
#define OFF_BL    110690304L                 // 2 KB   f32 [512]
#define OFF_H     110692352L                 // 2 MB   bf16 [4][2][128][1024]
#define OFF_C     112789504L                 // 2 MB   f32  [4][128][1024]
#define OFF_LOG   114886656L                 // 256 KB f32  [128][512]

extern "C" void kernel_launch(void* const* d_in, const int* in_sizes, int n_in,
                              void* d_out, int out_size, void* d_ws, size_t ws_size,
                              hipStream_t stream) {
  const float* x     = (const float*)d_in[0];
  const float* W_ih0 = (const float*)d_in[1];
  const float* W_hh0 = (const float*)d_in[2];
  const float* b_ih0 = (const float*)d_in[3];
  const float* b_hh0 = (const float*)d_in[4];
  const float* W_ihr = (const float*)d_in[5];
  const float* W_hhr = (const float*)d_in[6];
  const float* b_ihr = (const float*)d_in[7];
  const float* b_hhr = (const float*)d_in[8];
  const float* W_lin = (const float*)d_in[9];
  const float* b_lin = (const float*)d_in[10];

  char* ws = (char*)d_ws;
  __bf16* xbf   = (__bf16*)(ws + OFF_XBF);
  __bf16* W0    = (__bf16*)(ws + OFF_W0);
  __bf16* WR    = (__bf16*)(ws + OFF_WR);
  __bf16* WL    = (__bf16*)(ws + OFF_WL);
  float*  bias0 = (float*)(ws + OFF_B0);
  float*  biasr = (float*)(ws + OFF_BR);
  float*  biasl = (float*)(ws + OFF_BL);
  __bf16* hbuf  = (__bf16*)(ws + OFF_H);
  float*  cbuf  = (float*)(ws + OFF_C);
  float*  logit = (float*)(ws + OFF_LOG);

  auto Hp = [&](int layer, int par) { return hbuf + ((long)layer * 2 + par) * B_ * H_; };
  auto Cp = [&](int layer) { return cbuf + (long)layer * B_ * H_; };

  // ---- prep (deterministic each call; graph-safe) ----
  pack_x_kernel<<<8192, 256, 0, stream>>>(x, xbf);
  pack_weights_kernel<<<8192, 256, 0, stream>>>(W_ih0, D_, W_hh0, H_, W0, 4096L * 1536);
  for (int i = 0; i < 3; ++i)
    pack_weights_kernel<<<8192, 256, 0, stream>>>(
        W_ihr + (long)i * 4096 * 1536, 1536, W_hhr + (long)i * 4096 * 1024, 1024,
        WR + (long)i * 4096 * 2560, 4096L * 2560);
  pack_weights_kernel<<<2048, 256, 0, stream>>>(W_lin, 1536, W_lin, 0, WL, 512L * 1536);
  pack_bias_kernel<<<16, 256, 0, stream>>>(b_ih0, b_hh0, bias0, 4096, 1);
  for (int i = 0; i < 3; ++i)
    pack_bias_kernel<<<16, 256, 0, stream>>>(b_ihr + (long)i * 4096, b_hhr + (long)i * 4096,
                                             biasr + (long)i * 4096, 4096, 1);
  pack_bias_kernel<<<2, 256, 0, stream>>>(b_lin, b_lin, biasl, 512, 0);
  zero_state_kernel<<<2048, 256, 0, stream>>>(hbuf, cbuf, 4L * 2 * B_ * H_, 4L * B_ * H_);

  // ---- sequential timestep loop (stream-ordered; capturable) ----
  for (int t = 0; t < T_; ++t) {
    const __bf16* xt = xbf + (long)t * B_ * D_;
    const int p = t & 1, q = p ^ 1;

    // layer 0: K = 512(x) + 1024(h0 prev-t)
    lstm_cell_kernel<<<32, 256, 0, stream>>>(xt, Hp(0, q), Hp(0, q), W0, 1536, 1536,
                                             bias0, Cp(0), Hp(0, p));
    // layers 1..3: K = 512(x) + 1024(h_{l-1} cur-t) + 1024(h_l prev-t)
    for (int l = 1; l < 4; ++l)
      lstm_cell_kernel<<<32, 256, 0, stream>>>(
          xt, Hp(l - 1, p), Hp(l, q), WR + (long)(l - 1) * 4096 * 2560, 2560, 1536,
          biasr + (long)(l - 1) * 4096, Cp(l), Hp(l, p));

    head_kernel<<<16, 256, 0, stream>>>(xt, Hp(3, p), WL, biasl, logit);
    softmax_kernel<<<128, 256, 0, stream>>>(logit, (float*)d_out, t);
  }
}
